// TopKGating_63848983823106
// MI455X (gfx1250) — matrix operand order repaired and checked by reading, not compiled, not run on hardware
//
#include <hip/hip_runtime.h>

typedef __attribute__((ext_vector_type(16))) __bf16 v16bf;
typedef __attribute__((ext_vector_type(8)))  float  v8f;

#define T_TOKENS  16384
#define D_MODEL   2048
#define N_EXP     64
#define KCHUNK    128
#define WSTRIDE   136     // bf16 elems per LDS row (272 B rows: 16B-aligned)
#define SC_STRIDE 68      // padded float stride for score transpose (272 B rows)

// Pack two f32 into packed bf16 (truncation) with a single V_PERM_B32:
// result = { hi16(y), hi16(x) }.  sel bytes 0-3 pick from S1(x), 4-7 from S0(y).
static __device__ __forceinline__ unsigned pk2bf(float x, float y) {
#if __has_builtin(__builtin_amdgcn_perm)
  return __builtin_amdgcn_perm(__builtin_bit_cast(unsigned, y),
                               __builtin_bit_cast(unsigned, x),
                               0x07060302u);
#else
  unsigned ux = __builtin_bit_cast(unsigned, x);
  unsigned uy = __builtin_bit_cast(unsigned, y);
  return (ux >> 16) | (uy & 0xFFFF0000u);
#endif
}

union BfVec { unsigned u[8]; uint4 q[2]; v16bf v; };

__global__ __launch_bounds__(256) void topk_gate_kernel(
    const float* __restrict__ x, const float* __restrict__ W,
    const float* __restrict__ bias, float* __restrict__ out_vals,
    float* __restrict__ out_idx, float* __restrict__ out_scores)
{
  __shared__ __attribute__((aligned(16))) unsigned short wlds[N_EXP * WSTRIDE];
  __shared__ float bias_lds[N_EXP];
  __shared__ __attribute__((aligned(16))) float sc[8 * 16 * SC_STRIDE];

  const int tid   = threadIdx.x;
  const int w     = tid >> 5;          // wave in block (0..7)
  const int l     = tid & 31;          // lane
  const int ln    = l & 15;
  const int khalf = (l >> 4) << 3;     // A-matrix half-select: 0 or 8
  const int kbh   = (l >> 4) << 3;     // B-matrix half-select: 0 or 8
  const int m0    = (blockIdx.x * 8 + w) * 16;   // 16 tokens per wave
  const long mrow = (long)(m0 + ln);

  if (tid < N_EXP) bias_lds[tid] = bias[tid];

  v8f acc0 = {}, acc1 = {}, acc2 = {}, acc3 = {};

  const int e_st = tid >> 2;           // staging: expert row per thread
  const int seg  = (tid & 3) << 5;     // 32-element segment within 128-col chunk

  for (int k0 = 0; k0 < D_MODEL; k0 += KCHUNK) {
    __syncthreads();
    // --- cooperative stage: W[:, k0:k0+128] -> bf16 LDS (one convert per block) ---
    {
      const float4* src = (const float4*)(W + (long)e_st * D_MODEL + k0 + seg);
      unsigned* dst = (unsigned*)&wlds[e_st * WSTRIDE + seg];
      #pragma unroll
      for (int j = 0; j < 8; ++j) {
        float4 f = src[j];
        dst[2 * j]     = pk2bf(f.x, f.y);
        dst[2 * j + 1] = pk2bf(f.z, f.w);
      }
    }
    __syncthreads();

    #pragma unroll
    for (int sub = 0; sub < 4; ++sub) {
      const int kk = k0 + sub * 32;
      // --- A tile: 16 tokens x 32 k, bf16, per-ISA A layout ---
      BfVec A;
      {
        const float* px = x + mrow * D_MODEL + kk + khalf;
        float4 f0 = ((const float4*)px)[0];
        float4 f1 = ((const float4*)px)[1];
        float4 f2 = ((const float4*)(px + 16))[0];
        float4 f3 = ((const float4*)(px + 16))[1];
        A.u[0] = pk2bf(f0.x, f0.y);  A.u[1] = pk2bf(f0.z, f0.w);
        A.u[2] = pk2bf(f1.x, f1.y);  A.u[3] = pk2bf(f1.z, f1.w);
        A.u[4] = pk2bf(f2.x, f2.y);  A.u[5] = pk2bf(f2.z, f2.w);
        A.u[6] = pk2bf(f3.x, f3.y);  A.u[7] = pk2bf(f3.z, f3.w);
      }
      const int kb = sub * 32 + kbh;   // B: K 8h..8h+7 and 16+8h..16+8h+7 per lane
      // --- load all four B tiles, then 4 back-to-back WMMAs ---
      BfVec B0, B1, B2, B3;
      {
        const uint4* p0 = (const uint4*)&wlds[(ln)      * WSTRIDE + kb];
        const uint4* p1 = (const uint4*)&wlds[(16 + ln) * WSTRIDE + kb];
        const uint4* p2 = (const uint4*)&wlds[(32 + ln) * WSTRIDE + kb];
        const uint4* p3 = (const uint4*)&wlds[(48 + ln) * WSTRIDE + kb];
        B0.q[0] = p0[0]; B0.q[1] = p0[2];
        B1.q[0] = p1[0]; B1.q[1] = p1[2];
        B2.q[0] = p2[0]; B2.q[1] = p2[2];
        B3.q[0] = p3[0]; B3.q[1] = p3[2];
      }
      acc0 = __builtin_amdgcn_wmma_f32_16x16x32_bf16(false, A.v, false, B0.v, (short)0, acc0, false, false);
      acc1 = __builtin_amdgcn_wmma_f32_16x16x32_bf16(false, A.v, false, B1.v, (short)0, acc1, false, false);
      acc2 = __builtin_amdgcn_wmma_f32_16x16x32_bf16(false, A.v, false, B2.v, (short)0, acc2, false, false);
      acc3 = __builtin_amdgcn_wmma_f32_16x16x32_bf16(false, A.v, false, B3.v, (short)0, acc3, false, false);
    }
  }

  // --- transpose accumulators through LDS (per-wave region, same-wave LDS is in-order) ---
  const int mb = (l >> 4) << 3;   // rows 0-7 from lanes 0-15, rows 8-15 from lanes 16-31
  {
    float* scw = &sc[(w * 16 + mb) * SC_STRIDE];
    #pragma unroll
    for (int r = 0; r < 8; ++r) {
      float* row = scw + r * SC_STRIDE;
      row[ln]      = acc0[r];
      row[16 + ln] = acc1[r];
      row[32 + ln] = acc2[r];
      row[48 + ln] = acc3[r];
    }
  }

  // --- per-token bias + softmax + top-2 (lanes 0-15, one token each) ---
  if (l < 16) {
    const int token = m0 + l;
    float* row = &sc[(w * 16 + l) * SC_STRIDE];
    float mx = -3.4e38f;
    #pragma unroll 4
    for (int e = 0; e < N_EXP; ++e) {
      float v = row[e] + bias_lds[e];
      row[e] = v;
      mx = fmaxf(mx, v);
    }
    float s = 0.f;
    #pragma unroll 4
    for (int e = 0; e < N_EXP; ++e) {
      float p = __expf(row[e] - mx);
      row[e] = p;
      s += p;
    }
    const float inv = 1.f / s;
    float b1 = -1.f, b2 = -1.f;
    int   i1 = 0,    i2 = 0;
    float4* outs = (float4*)(out_scores + (long)token * N_EXP);
    for (int e4 = 0; e4 < N_EXP / 4; ++e4) {
      float4 pv = ((const float4*)row)[e4];
      pv.x *= inv; pv.y *= inv; pv.z *= inv; pv.w *= inv;
      outs[e4] = pv;
      const float pe[4] = {pv.x, pv.y, pv.z, pv.w};
      #pragma unroll
      for (int j = 0; j < 4; ++j) {
        const int e = e4 * 4 + j;
        const float p = pe[j];
        if (p > b1)      { b2 = b1; i2 = i1; b1 = p; i1 = e; }
        else if (p > b2) { b2 = p;  i2 = e; }
      }
    }
    out_vals[token * 2]     = b1;
    out_vals[token * 2 + 1] = b2;
    out_idx[token * 2]      = (float)i1;
    out_idx[token * 2 + 1]  = (float)i2;
  }
}

extern "C" void kernel_launch(void* const* d_in, const int* in_sizes, int n_in,
                              void* d_out, int out_size, void* d_ws, size_t ws_size,
                              hipStream_t stream) {
  (void)in_sizes; (void)n_in; (void)out_size; (void)d_ws; (void)ws_size;
  const float* x    = (const float*)d_in[0];
  const float* W    = (const float*)d_in[1];
  const float* bias = (const float*)d_in[2];
  float* out        = (float*)d_out;
  float* out_vals   = out;                      // [16384, 2]
  float* out_idx    = out + T_TOKENS * 2;       // [16384, 2] (indices as float)
  float* out_scores = out + T_TOKENS * 4;       // [16384, 64]

  dim3 grid(T_TOKENS / 128);   // 8 waves/block * 16 tokens/wave
  topk_gate_kernel<<<grid, 256, 0, stream>>>(x, W, bias, out_vals, out_idx, out_scores);
}
